// GraphModulatedAttention_31928786878550
// MI455X (gfx1250) — matrix operand order repaired and checked by reading, not compiled, not run on hardware
//
#include <hip/hip_runtime.h>
#include <hip/hip_bf16.h>

// ---------------------------------------------------------------------------
// GraphModulatedAttention forward for MI455X (gfx1250, wave32, WMMA 16x16x32 f16)
// V is stored transposed so every WMMA operand load in the flash loop is a
// contiguous 32-byte run per lane (vectorizes to global_load_b128).
// ---------------------------------------------------------------------------
#define B_   8
#define C_   256
#define HW_  1024
#define NH_  8
#define HD_  32
#define NN_  102          // top-k nodes
#define GD_  64           // graph dim

typedef __attribute__((ext_vector_type(16))) _Float16 v16h;
typedef __attribute__((ext_vector_type(8)))  float    v8f;

// ---------------- WMMA helpers (contiguous k = 16*(lane/16) + i in both operands, wave32) -------

// A-matrix 16x32 f16: lanes 0-15 -> M=lane, K=0..15 ; lanes 16-31 -> M=lane-16, K=16..31
__device__ __forceinline__ v16h load_a16(const _Float16* A, int lda, int lane) {
  int m  = lane & 15;
  int kb = (lane < 16) ? 0 : 16;
  v16h a;
#pragma unroll
  for (int i = 0; i < 16; ++i) a[i] = A[m * lda + kb + i];          // K = kb + i
  return a;
}

// B-matrix 32x16 where logical B[k][n] = W[n][k] (row-major W, contiguous in k)
__device__ __forceinline__ v16h load_b_wt(const _Float16* W, int ldw, int lane) {
  int n  = lane & 15;
  int kb = (lane < 16) ? 0 : 16;
  v16h b;
#pragma unroll
  for (int i = 0; i < 16; ++i) b[i] = W[n * ldw + kb + i];
  return b;
}

__device__ __forceinline__ v8f wmma16(v16h a, v16h b, v8f c) {
  return __builtin_amdgcn_wmma_f32_16x16x32_f16(false, a, false, b, (short)0, c, false, false);
}

// ---------------- prep kernels ---------------------------------------------

__global__ void k_importance(const float* __restrict__ x, float* __restrict__ imp) {
  int tid = blockIdx.x * blockDim.x + threadIdx.x;       // B*HW
  if (tid >= B_ * HW_) return;
  int b = tid >> 10, p = tid & 1023;
  const float* xp = x + (size_t)b * C_ * HW_ + p;
  float s = 0.f, ss = 0.f;
  for (int c = 0; c < C_; ++c) { float v = xp[c * HW_]; s += v; ss += v * v; }
  float mean = s * (1.0f / C_);
  imp[tid] = (ss - s * mean) * (1.0f / (C_ - 1));        // unbiased var
}

__global__ void k_xh(const float* __restrict__ x, _Float16* __restrict__ xh) {
  size_t i = (size_t)blockIdx.x * 256 + threadIdx.x;     // B*HW*C elements
  int c = i & 255; int p = (i >> 8) & 1023; int b = i >> 18;
  xh[i] = (_Float16)x[((size_t)b * C_ + c) * HW_ + p];
}

__global__ void k_f2h(const float* __restrict__ s, _Float16* __restrict__ d, int n) {
  int i = blockIdx.x * 256 + threadIdx.x;
  if (i < n) d[i] = (_Float16)s[i];
}

// one block (1024 thr) per batch: iterative argmax top-102, lower index wins ties
__global__ void k_topk(const float* __restrict__ imp, int* __restrict__ top_idx,
                       int* __restrict__ nodepos) {
  __shared__ float sv[1024];
  __shared__ float rv[1024];
  __shared__ int   ri[1024];
  int b = blockIdx.x, t = threadIdx.x;
  sv[t] = imp[b * HW_ + t];
  nodepos[b * HW_ + t] = -1;
  __syncthreads();
  for (int it = 0; it < NN_; ++it) {
    rv[t] = sv[t]; ri[t] = t;
    __syncthreads();
    for (int s = 512; s > 0; s >>= 1) {
      if (t < s) {
        float v2 = rv[t + s]; int i2 = ri[t + s];
        if (v2 > rv[t] || (v2 == rv[t] && i2 < ri[t])) { rv[t] = v2; ri[t] = i2; }
      }
      __syncthreads();
    }
    if (t == 0) {
      int sel = ri[0];
      top_idx[b * NN_ + it] = sel;
      nodepos[b * HW_ + sel] = it;
      sv[sel] = -3.0e38f;
    }
    __syncthreads();
  }
}

// one block (256 thr) per (b,i): gather feats + l2-normalize
__global__ void k_feats(const float* __restrict__ x, const int* __restrict__ top_idx,
                        float* __restrict__ feats, float* __restrict__ fn) {
  __shared__ float red[256];
  int bi = blockIdx.x;                 // b*NN + i
  int b = bi / NN_;
  int c = threadIdx.x;
  int p = top_idx[bi];
  float v = x[((size_t)b * C_ + c) * HW_ + p];
  feats[(size_t)bi * C_ + c] = v;
  red[c] = v * v;
  __syncthreads();
  for (int s = 128; s > 0; s >>= 1) { if (c < s) red[c] += red[c + s]; __syncthreads(); }
  float nrm = fmaxf(sqrtf(red[0]), 1e-12f);
  fn[(size_t)bi * C_ + c] = v / nrm;
}

__global__ void k_adj(const float* __restrict__ fn, unsigned char* __restrict__ adj) {
  int t = blockIdx.x * 256 + threadIdx.x;
  if (t >= B_ * NN_ * NN_) return;
  int j = t % NN_; int i = (t / NN_) % NN_; int b = t / (NN_ * NN_);
  const float* a = fn + ((size_t)b * NN_ + i) * C_;
  const float* c = fn + ((size_t)b * NN_ + j) * C_;
  float s = 0.f;
  for (int k = 0; k < C_; ++k) s += a[k] * c[k];
  adj[t] = (unsigned char)((s > 0.6f) || (i == j));
}

// full 2-layer GAT + graph_to_attn per batch (one block, 256 threads)
__global__ void k_gat(const float* __restrict__ feats, const unsigned char* __restrict__ adjg,
                      const float* g0W, const float* g0as, const float* g0ad, const float* g0b,
                      const float* g1W, const float* g1as, const float* g1ad, const float* g1b,
                      const float* wg2a, const float* bg2a, float* __restrict__ gw) {
  __shared__ float hbuf[NN_ * GD_];
  __shared__ float gbuf[NN_ * GD_];
  __shared__ float asr[NN_], ads[NN_], rmx[NN_], rdn[NN_];
  int b = blockIdx.x, t = threadIdx.x;
  const unsigned char* A = adjg + b * NN_ * NN_;

  // ---- layer 0: h = feats @ g0W (102x256 @ 256x64) ----
  for (int e = t; e < NN_ * GD_; e += 256) {
    int i = e >> 6, d = e & 63;
    const float* f = feats + ((size_t)b * NN_ + i) * C_;
    float s = 0.f;
    for (int k = 0; k < C_; ++k) s += f[k] * g0W[k * GD_ + d];
    hbuf[e] = s;
  }
  __syncthreads();
  if (t < NN_) {
    float s1 = 0.f, s2 = 0.f;
    for (int d = 0; d < GD_; ++d) { s1 += hbuf[t * GD_ + d] * g0as[d]; s2 += hbuf[t * GD_ + d] * g0ad[d]; }
    asr[t] = s1; ads[t] = s2;
  }
  __syncthreads();
  if (t < NN_) {                                 // softmax stats over sources j (mask.T)
    float mx = -3e38f;
    for (int j = 0; j < NN_; ++j) if (A[j * NN_ + t]) {
      float z = ads[t] + asr[j]; z = (z > 0.f) ? z : 0.2f * z;
      mx = fmaxf(mx, z);
    }
    float dn = 0.f;
    for (int j = 0; j < NN_; ++j) if (A[j * NN_ + t]) {
      float z = ads[t] + asr[j]; z = (z > 0.f) ? z : 0.2f * z;
      dn += __expf(z - mx);
    }
    rmx[t] = mx; rdn[t] = dn;
  }
  __syncthreads();
  for (int e = t; e < NN_ * GD_; e += 256) {     // aggregate + bias + relu
    int i = e >> 6, d = e & 63;
    float s = 0.f;
    for (int j = 0; j < NN_; ++j) if (A[j * NN_ + i]) {
      float z = ads[i] + asr[j]; z = (z > 0.f) ? z : 0.2f * z;
      s += __expf(z - rmx[i]) * hbuf[j * GD_ + d];
    }
    float o = s / rdn[i] + g0b[d];
    gbuf[e] = (o > 0.f) ? o : 0.f;
  }
  __syncthreads();

  // ---- layer 1: h = g @ g1W (102x64 @ 64x64) ----
  for (int e = t; e < NN_ * GD_; e += 256) {
    int i = e >> 6, d = e & 63;
    float s = 0.f;
    for (int k = 0; k < GD_; ++k) s += gbuf[i * GD_ + k] * g1W[k * GD_ + d];
    hbuf[e] = s;
  }
  __syncthreads();
  if (t < NN_) {
    float s1 = 0.f, s2 = 0.f;
    for (int d = 0; d < GD_; ++d) { s1 += hbuf[t * GD_ + d] * g1as[d]; s2 += hbuf[t * GD_ + d] * g1ad[d]; }
    asr[t] = s1; ads[t] = s2;
  }
  __syncthreads();
  if (t < NN_) {
    float mx = -3e38f;
    for (int j = 0; j < NN_; ++j) if (A[j * NN_ + t]) {
      float z = ads[t] + asr[j]; z = (z > 0.f) ? z : 0.2f * z;
      mx = fmaxf(mx, z);
    }
    float dn = 0.f;
    for (int j = 0; j < NN_; ++j) if (A[j * NN_ + t]) {
      float z = ads[t] + asr[j]; z = (z > 0.f) ? z : 0.2f * z;
      dn += __expf(z - mx);
    }
    rmx[t] = mx; rdn[t] = dn;
  }
  __syncthreads();
  for (int e = t; e < NN_ * GD_; e += 256) {
    int i = e >> 6, d = e & 63;
    float s = 0.f;
    for (int j = 0; j < NN_; ++j) if (A[j * NN_ + i]) {
      float z = ads[i] + asr[j]; z = (z > 0.f) ? z : 0.2f * z;
      s += __expf(z - rmx[i]) * hbuf[j * GD_ + d];
    }
    float o = s / rdn[i] + g1b[d];
    gbuf[e] = (o > 0.f) ? o : 0.f;
  }
  __syncthreads();

  // ---- graph_to_attn: gw = sigmoid(g @ wg2a^T + bg2a) ----
  for (int e = t; e < NN_ * NH_; e += 256) {
    int i = e >> 3, hh = e & 7;
    float s = bg2a[hh];
    for (int d = 0; d < GD_; ++d) s += gbuf[i * GD_ + d] * wg2a[hh * GD_ + d];
    gw[((size_t)b * NN_ + i) * NH_ + hh] = 1.0f / (1.0f + __expf(-s));
  }
}

// ---------------- WMMA GEMM kernels ----------------------------------------

// qkv = x_seq @ w_qkv^T ;
// Q,K scatter into [sel][B][NH][HW][HD] f16; V scatter TRANSPOSED into
// section 2 as [B][NH][HD][HW] f16 so the flash P@V operand is contiguous.
__global__ void k_qkv_gemm(const _Float16* __restrict__ xh, const _Float16* __restrict__ wq,
                           _Float16* __restrict__ qkv) {
  int lane = threadIdx.x & 31, wave = threadIdx.x >> 5;
  int tile = blockIdx.x * 4 + wave;          // (B*HW/16) * (3C/16) = 512*48
  int mt = tile / 48, nt = tile % 48;
  const _Float16* Ab = xh + (size_t)(mt * 16) * C_;
  const _Float16* Wb = wq + (size_t)(nt * 16) * C_;
  v8f acc = {};
#pragma unroll
  for (int k0 = 0; k0 < C_; k0 += 32) {
    v16h a = load_a16(Ab + k0, C_, lane);
    v16h b = load_b_wt(Wb + k0, C_, lane);
    acc = wmma16(a, b, acc);
  }
  int n = nt * 16 + (lane & 15);             // 0..767
  int sel = n >> 8, c = n & 255;
  int h = c >> 5, d = c & 31;
  int mbase = mt * 16 + ((lane < 16) ? 0 : 8);
  const size_t VSEC = (size_t)2 * B_ * NH_ * HW_ * HD_;
#pragma unroll
  for (int r = 0; r < 8; ++r) {
    int gm = mbase + r;
    int bb = gm >> 10, p = gm & 1023;
    size_t dst;
    if (sel < 2)
      dst = ((((size_t)sel * B_ + bb) * NH_ + h) * HW_ + p) * HD_ + d;   // [sel][b][h][p][d]
    else
      dst = VSEC + (((size_t)bb * NH_ + h) * HD_ + d) * HW_ + p;         // Vt: [b][h][d][p]
    qkv[dst] = (_Float16)acc[r];
  }
}

// fused flash attention + graph modulation; one wave per (b,h,16-query tile)
__global__ void k_flash(const _Float16* __restrict__ qkv, const int* __restrict__ nodepos,
                        const unsigned char* __restrict__ adj, const float* __restrict__ gw,
                        _Float16* __restrict__ ao) {
  __shared__ _Float16 pl[4][16 * 32];
  int lane = threadIdx.x & 31, wave = threadIdx.x >> 5;
  int tile = blockIdx.x * 4 + wave;          // B*NH*(HW/16) = 4096
  int qt = tile & 63;
  int h  = (tile >> 6) & 7;
  int b  = tile >> 9;

  const _Float16* Q  = qkv + (((size_t)0 * B_ + b) * NH_ + h) * HW_ * HD_;
  const _Float16* K  = qkv + (((size_t)1 * B_ + b) * NH_ + h) * HW_ * HD_;
  const _Float16* Vt = qkv + (size_t)2 * B_ * NH_ * HW_ * HD_
                           + ((size_t)b * NH_ + h) * HD_ * HW_;          // [HD][HW]
  int qbase = qt * 16;
  int mloc  = (lane < 16) ? 0 : 8;

  v16h qa = load_a16(Q + (size_t)qbase * HD_, HD_, lane);

  int   nrow[8]; float gwrow[8];
#pragma unroll
  for (int r = 0; r < 8; ++r) {
    int qp = qbase + mloc + r;
    int nd = nodepos[b * HW_ + qp];
    nrow[r] = nd;
    gwrow[r] = (nd >= 0) ? gw[((size_t)b * NN_ + nd) * NH_ + h] : 0.f;
  }

  float mst[8], lst[8];
  v8f o0 = {}, o1 = {};
#pragma unroll
  for (int r = 0; r < 8; ++r) { mst[r] = -3.0e38f; lst[r] = 0.f; }
  const float scale = 0.17677669529663687f;  // hd^-0.5

  for (int kb = 0; kb < HW_; kb += 32) {
    v16h b0 = load_b_wt(K + (size_t)kb * HD_, HD_, lane);
    v16h b1 = load_b_wt(K + (size_t)(kb + 16) * HD_, HD_, lane);
    v8f zc = {};
    v8f s0 = wmma16(qa, b0, zc);
    v8f s1 = wmma16(qa, b1, zc);

    // graph modulation columns for this lane
    int kp0 = kb + (lane & 15), kp1 = kp0 + 16;
    int nc0 = nodepos[b * HW_ + kp0];
    int nc1 = nodepos[b * HW_ + kp1];
    float gc0 = 0.f, gc1 = 0.f;
    if (nc0 >= 0) gc0 = gw[((size_t)b * NN_ + nc0) * NH_ + h];
    if (nc1 >= 0) gc1 = gw[((size_t)b * NN_ + nc1) * NH_ + h];

    float p0[8], p1[8];
#pragma unroll
    for (int r = 0; r < 8; ++r) {
      float v0 = s0[r] * scale, v1 = s1[r] * scale;
      if (nrow[r] >= 0) {
        const unsigned char* arow = adj + ((size_t)b * NN_ + nrow[r]) * NN_;
        if (nc0 >= 0 && arow[nc0]) v0 += gwrow[r] * gc0;
        if (nc1 >= 0 && arow[nc1]) v1 += gwrow[r] * gc1;
      }
      float rm = fmaxf(v0, v1);
#pragma unroll
      for (int off = 1; off < 16; off <<= 1) rm = fmaxf(rm, __shfl_xor(rm, off, 32));
      float mnew = fmaxf(mst[r], rm);
      p0[r] = __expf(v0 - mnew);
      p1[r] = __expf(v1 - mnew);
      float rs = p0[r] + p1[r];
#pragma unroll
      for (int off = 1; off < 16; off <<= 1) rs += __shfl_xor(rs, off, 32);
      float corr = __expf(mst[r] - mnew);
      lst[r] = lst[r] * corr + rs;
      mst[r] = mnew;
      o0[r] *= corr;
      o1[r] *= corr;
    }

    // stage P (C-layout f32) -> LDS row-major f16 -> A-layout (per-wave LDS, in-order DS)
    _Float16* pw = pl[wave];
    int n = lane & 15;
#pragma unroll
    for (int r = 0; r < 8; ++r) {
      int m = mloc + r;
      pw[m * 32 + n]      = (_Float16)p0[r];
      pw[m * 32 + 16 + n] = (_Float16)p1[r];
    }
    v16h pa  = load_a16(pw, 32, lane);
    // Vt is [HD][HW]: B[k=key][n=hd] = Vt[hd0+n][kb+k]  -> contiguous along k
    v16h vb0 = load_b_wt(Vt + kb,             HW_, lane);   // hd 0..15
    v16h vb1 = load_b_wt(Vt + 16 * HW_ + kb,  HW_, lane);   // hd 16..31
    o0 = wmma16(pa, vb0, o0);
    o1 = wmma16(pa, vb1, o1);
  }

  // epilogue: O /= l ; store f16 [B, HW, C] at channel h*32 + d
  int n = lane & 15;
#pragma unroll
  for (int r = 0; r < 8; ++r) {
    int qp = qbase + mloc + r;
    float inv = 1.0f / lst[r];
    size_t base = ((size_t)b * HW_ + qp) * C_ + h * HD_;
    ao[base + n]      = (_Float16)(o0[r] * inv);
    ao[base + 16 + n] = (_Float16)(o1[r] * inv);
  }
}

// out = attn_out @ w_proj^T + b_proj, stored transposed as [B, C, HW] f32
__global__ void k_proj_gemm(const _Float16* __restrict__ ao, const _Float16* __restrict__ wp,
                            const float* __restrict__ bias, float* __restrict__ out) {
  int lane = threadIdx.x & 31, wave = threadIdx.x >> 5;
  int tile = blockIdx.x * 4 + wave;          // 512 * 16
  int mt = tile / 16, nt = tile % 16;
  const _Float16* Ab = ao + (size_t)(mt * 16) * C_;
  const _Float16* Wb = wp + (size_t)(nt * 16) * C_;
  v8f acc = {};
#pragma unroll
  for (int k0 = 0; k0 < C_; k0 += 32) {
    v16h a = load_a16(Ab + k0, C_, lane);
    v16h b = load_b_wt(Wb + k0, C_, lane);
    acc = wmma16(a, b, acc);
  }
  int n = nt * 16 + (lane & 15);
  float bv = bias[n];
  int mbase = mt * 16 + ((lane < 16) ? 0 : 8);
#pragma unroll
  for (int r = 0; r < 8; ++r) {
    int gm = mbase + r;
    int bb = gm >> 10, p = gm & 1023;
    out[((size_t)bb * C_ + n) * HW_ + p] = acc[r] + bv;
  }
}

// ---------------------------------------------------------------------------

static inline size_t align_up(size_t v, size_t a) { return (v + a - 1) & ~(a - 1); }

extern "C" void kernel_launch(void* const* d_in, const int* in_sizes, int n_in,
                              void* d_out, int out_size, void* d_ws, size_t ws_size,
                              hipStream_t stream) {
  (void)in_sizes; (void)n_in; (void)out_size; (void)ws_size;
  const float* x      = (const float*)d_in[0];
  const float* w_qkv  = (const float*)d_in[1];
  const float* w_proj = (const float*)d_in[2];
  const float* b_proj = (const float*)d_in[3];
  const float* g0W    = (const float*)d_in[4];
  const float* g0as   = (const float*)d_in[5];
  const float* g0ad   = (const float*)d_in[6];
  const float* g0b    = (const float*)d_in[7];
  const float* g1W    = (const float*)d_in[8];
  const float* g1as   = (const float*)d_in[9];
  const float* g1ad   = (const float*)d_in[10];
  const float* g1b    = (const float*)d_in[11];
  const float* wg2a   = (const float*)d_in[12];
  const float* bg2a   = (const float*)d_in[13];
  float* out = (float*)d_out;

  char* ws = (char*)d_ws;
  size_t off = 0;
  auto alloc = [&](size_t bytes) { size_t o = off; off = align_up(off + bytes, 256); return (void*)(ws + o); };

  float*         imp     = (float*)        alloc(B_ * HW_ * 4);
  int*           top_idx = (int*)          alloc(B_ * NN_ * 4);
  int*           nodepos = (int*)          alloc(B_ * HW_ * 4);
  float*         feats   = (float*)        alloc((size_t)B_ * NN_ * C_ * 4);
  float*         fn      = (float*)        alloc((size_t)B_ * NN_ * C_ * 4);
  unsigned char* adj     = (unsigned char*)alloc((size_t)B_ * NN_ * NN_);
  float*         gw      = (float*)        alloc((size_t)B_ * NN_ * NH_ * 4);
  _Float16*      xh      = (_Float16*)     alloc((size_t)B_ * HW_ * C_ * 2);
  _Float16*      wqh     = (_Float16*)     alloc((size_t)3 * C_ * C_ * 2);
  _Float16*      wph     = (_Float16*)     alloc((size_t)C_ * C_ * 2);
  _Float16*      qkv     = (_Float16*)     alloc((size_t)3 * B_ * NH_ * HW_ * HD_ * 2);
  _Float16*      ao      = (_Float16*)     alloc((size_t)B_ * HW_ * C_ * 2);

  // ---- prep ----
  k_importance<<<(B_ * HW_) / 256, 256, 0, stream>>>(x, imp);
  k_xh<<<(B_ * HW_ * C_) / 256, 256, 0, stream>>>(x, xh);
  k_f2h<<<(3 * C_ * C_) / 256, 256, 0, stream>>>(w_qkv, wqh, 3 * C_ * C_);
  k_f2h<<<(C_ * C_) / 256, 256, 0, stream>>>(w_proj, wph, C_ * C_);
  k_topk<<<B_, 1024, 0, stream>>>(imp, top_idx, nodepos);
  k_feats<<<B_ * NN_, 256, 0, stream>>>(x, top_idx, feats, fn);
  k_adj<<<(B_ * NN_ * NN_ + 255) / 256, 256, 0, stream>>>(fn, adj);
  k_gat<<<B_, 256, 0, stream>>>(feats, adj, g0W, g0as, g0ad, g0b,
                                g1W, g1as, g1ad, g1b, wg2a, bg2a, gw);

  // ---- WMMA pipeline ----
  k_qkv_gemm<<<(512 * 48) / 4, 128, 0, stream>>>(xh, wqh, qkv);
  k_flash<<<(B_ * NH_ * (HW_ / 16)) / 4, 128, 0, stream>>>(qkv, nodepos, adj, gw, ao);
  k_proj_gemm<<<(512 * 16) / 4, 128, 0, stream>>>(ao, wph, b_proj, out);
}
